// TabularAttentionModel_33337536152147
// MI455X (gfx1250) — compile-verified, matrix operand order checked by reading, not run on hardware
//
#include <hip/hip_runtime.h>
#include <hip/hip_bf16.h>
#include <math.h>

#define BB    8192
#define NFEAT 200
#define EPS   1e-5f

typedef __attribute__((ext_vector_type(16))) __bf16 v16bf;
typedef __attribute__((ext_vector_type(8)))  float  v8f;
typedef unsigned uv4 __attribute__((ext_vector_type(4)));
typedef unsigned uv8 __attribute__((ext_vector_type(8)));

// ---- workspace layout (float offsets) ----
#define ST_FEAT   0        // m_feat, rs_feat, m_featw, rs_featw
#define ST_CH_SUM 8        // 200 * [xs,xss,rs,rss,ns,nss]
#define ST_CH_FIN 1208     // 200 * [mx,rsx,mr,rsr,mn,rsn]
#define ST_XW_SUM 2408     // 26 * [s,ss]
#define ST_XW_FIN 2460     // 26 * [m,rs]
#define ST_PL_SUM 2512     // 24 * [s,ss]
#define ST_PL_FIN 2560
#define ST_Y0_SUM 2608     // 1000 * [s,ss]
#define ST_Y0_FIN 4608
#define ST_Y1_SUM 6608     // 1500 * [s,ss]
#define ST_Y1_FIN 9608
#define STATS_N   12608

#define OFF_RAW   16384
#define OFF_NOTU  (OFF_RAW  + 10 * NFEAT * BB)
#define OFF_LOGIT (OFF_NOTU + 10 * NFEAT * BB)
#define OFF_POOL  (OFF_LOGIT + NFEAT * BB)
#define OFF_Y0    (OFF_POOL + 24 * BB)
#define OFF_Y1    (OFF_Y0   + 1000 * BB)
#define OFF_Y0BF  (OFF_Y1   + 1500 * BB)            // bf16 8192 x 1024 (as 4,194,304 dwords)
#define OFF_W1S   (OFF_Y0BF + 512 * BB)             // 32*96*32*8 = 786,432 dwords
#define NTILES_K  32
#define NTILES_N  96

static __device__ __forceinline__ float warp_red(float v) {
  for (int o = 16; o > 0; o >>= 1) v += __shfl_down(v, o);
  return v;
}

// ---------------------------------------------------------------- zero stats
__global__ void k_zero(float* ws) {
  int i = blockIdx.x * 256 + threadIdx.x;
  if (i < STATS_N) ws[i] = 0.f;
}

// --------------------------------------------- feat / featw broadcast stats
__global__ __launch_bounds__(1024) void k_feat_stats(
    const int* __restrict__ xcat, const float* __restrict__ ef,
    const float* __restrict__ efw, float* __restrict__ ws) {
  int t = threadIdx.x;
  float s = 0, ss = 0, sw = 0, ssw = 0;
  for (int b = t; b < BB; b += 1024) {
    int fi = xcat[b * (NFEAT + 1) + NFEAT];
    float a = ef[fi * 2], c = ef[fi * 2 + 1];
    s  += a + c;  ss  += a * a + c * c;
    float d = efw[fi * 2], e = efw[fi * 2 + 1];
    sw += d + e;  ssw += d * d + e * e;
  }
  __shared__ float red[32][4];
  s = warp_red(s); ss = warp_red(ss); sw = warp_red(sw); ssw = warp_red(ssw);
  if ((t & 31) == 0) { int w = t >> 5; red[w][0] = s; red[w][1] = ss; red[w][2] = sw; red[w][3] = ssw; }
  __syncthreads();
  if (t == 0) {
    float S = 0, SS = 0, SW = 0, SSW = 0;
    for (int w = 0; w < 32; w++) { S += red[w][0]; SS += red[w][1]; SW += red[w][2]; SSW += red[w][3]; }
    float cnt = 2.f * BB;
    float m = S / cnt,  v = SS / cnt - m * m;
    ws[ST_FEAT + 0] = m; ws[ST_FEAT + 1] = rsqrtf(v + EPS);
    m = SW / cnt; v = SSW / cnt - m * m;
    ws[ST_FEAT + 2] = m; ws[ST_FEAT + 3] = rsqrtf(v + EPS);
  }
}

// ---------------- per-(b,f): both 3->50->10 MLPs + channel BN3 sum gather
#define RW1 0
#define RB1 150
#define RW2 200
#define RB2 700
#define NW1 710
#define NB1 860
#define NW2 910
#define NB2 1410
__global__ __launch_bounds__(256) void k_percol_mlp(
    const int* __restrict__ xcat, const float* __restrict__ xcont,
    const float* __restrict__ ef, const float* __restrict__ emb_cat,
    const float* __restrict__ rw1, const float* __restrict__ rb1,
    const float* __restrict__ rw2, const float* __restrict__ rb2,
    const float* __restrict__ nw1, const float* __restrict__ nb1,
    const float* __restrict__ nw2, const float* __restrict__ nb2,
    const float* __restrict__ gfeat, const float* __restrict__ bfeat,
    float* __restrict__ ws) {
  __shared__ float W[1420];
  int t = threadIdx.x;
  for (int i = t; i < 150; i += 256) { W[RW1 + i] = rw1[i]; W[NW1 + i] = nw1[i]; }
  for (int i = t; i < 50;  i += 256) { W[RB1 + i] = rb1[i]; W[NB1 + i] = nb1[i]; }
  for (int i = t; i < 500; i += 256) { W[RW2 + i] = rw2[i]; W[NW2 + i] = nw2[i]; }
  for (int i = t; i < 10;  i += 256) { W[RB2 + i] = rb2[i]; W[NB2 + i] = nb2[i]; }
  __syncthreads();
  int f = blockIdx.y;
  int b = blockIdx.x * 256 + t;
  float mf = ws[ST_FEAT + 0], rsf = ws[ST_FEAT + 1];
  int fi = xcat[b * (NFEAT + 1) + NFEAT];
  float gf = gfeat[f], bf = bfeat[f];
  float xf0 = (ef[fi * 2]     - mf) * rsf * gf + bf;
  float xf1 = (ef[fi * 2 + 1] - mf) * rsf * gf + bf;

  float outr[10], outn[10];
  float inr = xcont[b * 2 * NFEAT + f];
  float inn = xcont[b * 2 * NFEAT + NFEAT + f];
  for (int c = 0; c < 10; c++) { outr[c] = W[RB2 + c]; outn[c] = W[NB2 + c]; }
  for (int j = 0; j < 50; j++) {
    float hr = inr * W[RW1 + j] + xf0 * W[RW1 + 50 + j] + xf1 * W[RW1 + 100 + j] + W[RB1 + j];
    hr = fmaxf(hr, 0.f);
    float hn = inn * W[NW1 + j] + xf0 * W[NW1 + 50 + j] + xf1 * W[NW1 + 100 + j] + W[NB1 + j];
    hn = fmaxf(hn, 0.f);
    for (int c = 0; c < 10; c++) {
      outr[c] += hr * W[RW2 + j * 10 + c];
      outn[c] += hn * W[NW2 + j * 10 + c];
    }
  }
  float rs_ = 0, rss = 0, ns = 0, nss = 0;
  for (int c = 0; c < 10; c++) {
    ws[OFF_RAW  + (size_t)(f * 10 + c) * BB + b] = outr[c];
    ws[OFF_NOTU + (size_t)(f * 10 + c) * BB + b] = outn[c];
    rs_ += outr[c]; rss += outr[c] * outr[c];
    ns  += outn[c]; nss += outn[c] * outn[c];
  }
  int cat = xcat[b * (NFEAT + 1) + f];
  float xs = 0, xss = 0;
  for (int l = 0; l < 4; l++) { float e = emb_cat[cat * 4 + l]; xs += e; xss += e * e; }

  __shared__ float red[8][6];
  float v6[6] = {xs, xss, rs_, rss, ns, nss};
  for (int i = 0; i < 6; i++) {
    float r = warp_red(v6[i]);
    if ((t & 31) == 0) red[t >> 5][i] = r;
  }
  __syncthreads();
  if (t < 6) {
    float s = 0;
    for (int w = 0; w < 8; w++) s += red[w][t];
    atomicAdd(&ws[ST_CH_SUM + f * 6 + t], s);
  }
}

__global__ void k_ch_final(float* ws) {
  int f = threadIdx.x;
  if (f >= NFEAT) return;
  const float cnt[3] = {4.f * BB, 10.f * BB, 10.f * BB};
  for (int p = 0; p < 3; p++) {
    float s  = ws[ST_CH_SUM + f * 6 + p * 2];
    float ss = ws[ST_CH_SUM + f * 6 + p * 2 + 1];
    float m = s / cnt[p];
    float v = ss / cnt[p] - m * m;
    ws[ST_CH_FIN + f * 6 + p * 2]     = m;
    ws[ST_CH_FIN + f * 6 + p * 2 + 1] = rsqrtf(v + EPS);
  }
}

// ------------- helper: build the 26 BN3'd attention-input components
static __device__ __forceinline__ void build_xw(
    const int* xcat, const float* emb_cat, const float* efw,
    const float* g_cat, const float* b_cat, const float* g_fw, const float* b_fw,
    const float* g_raw, const float* b_raw, const float* g_notu, const float* b_notu,
    const float* ws, int b, int f, float* v) {
  const float* chf = ws + ST_CH_FIN + f * 6;
  float mx = chf[0], rsx = chf[1], mr = chf[2], rsr = chf[3], mn = chf[4], rsn = chf[5];
  int cat = xcat[b * (NFEAT + 1) + f];
  for (int l = 0; l < 4; l++)
    v[l] = (emb_cat[cat * 4 + l] - mx) * rsx * g_cat[f] + b_cat[f];
  int fi = xcat[b * (NFEAT + 1) + NFEAT];
  float mfw = ws[ST_FEAT + 2], rsfw = ws[ST_FEAT + 3];
  v[4] = (efw[fi * 2]     - mfw) * rsfw * g_fw[f] + b_fw[f];
  v[5] = (efw[fi * 2 + 1] - mfw) * rsfw * g_fw[f] + b_fw[f];
  for (int c = 0; c < 10; c++) {
    v[6 + c]  = (ws[OFF_RAW  + (size_t)(f * 10 + c) * BB + b] - mr) * rsr * g_raw[f]  + b_raw[f];
    v[16 + c] = (ws[OFF_NOTU + (size_t)(f * 10 + c) * BB + b] - mn) * rsn * g_notu[f] + b_notu[f];
  }
}

// ---------------------------------------------- xw BN2 statistics gather
__global__ __launch_bounds__(256) void k_xw_sums(
    const int* __restrict__ xcat, const float* __restrict__ emb_cat,
    const float* __restrict__ efw,
    const float* __restrict__ g_cat, const float* __restrict__ b_cat,
    const float* __restrict__ g_fw, const float* __restrict__ b_fw,
    const float* __restrict__ g_raw, const float* __restrict__ b_raw,
    const float* __restrict__ g_notu, const float* __restrict__ b_notu,
    float* __restrict__ ws) {
  int t = threadIdx.x;
  int f = blockIdx.y;
  int b = blockIdx.x * 256 + t;
  float v[26];
  build_xw(xcat, emb_cat, efw, g_cat, b_cat, g_fw, b_fw, g_raw, b_raw, g_notu, b_notu, ws, b, f, v);
  __shared__ float red[8][52];
  for (int i = 0; i < 26; i++) {
    float s  = warp_red(v[i]);
    float ss = warp_red(v[i] * v[i]);
    if ((t & 31) == 0) { red[t >> 5][i * 2] = s; red[t >> 5][i * 2 + 1] = ss; }
  }
  __syncthreads();
  if (t < 52) {
    float s = 0;
    for (int w = 0; w < 8; w++) s += red[w][t];
    atomicAdd(&ws[ST_XW_SUM + t], s);
  }
}

// ------------------------------------------------- attention logits
__global__ __launch_bounds__(256) void k_logits(
    const int* __restrict__ xcat, const float* __restrict__ emb_cat,
    const float* __restrict__ efw,
    const float* __restrict__ g_cat, const float* __restrict__ b_cat,
    const float* __restrict__ g_fw, const float* __restrict__ b_fw,
    const float* __restrict__ g_raw, const float* __restrict__ b_raw,
    const float* __restrict__ g_notu, const float* __restrict__ b_notu,
    const float* __restrict__ g_w, const float* __restrict__ b_w,
    const float* __restrict__ aw1, const float* __restrict__ ab1,
    const float* __restrict__ aw2, const float* __restrict__ ab2,
    float* __restrict__ ws) {
  int t = threadIdx.x;
  int f = blockIdx.y;
  int b = blockIdx.x * 256 + t;
  float v[26];
  build_xw(xcat, emb_cat, efw, g_cat, b_cat, g_fw, b_fw, g_raw, b_raw, g_notu, b_notu, ws, b, f, v);
  float xw[26];
  for (int i = 0; i < 26; i++)
    xw[i] = (v[i] - ws[ST_XW_FIN + i * 2]) * ws[ST_XW_FIN + i * 2 + 1] * g_w[i] + b_w[i];
  float logit = ab2[0];
  for (int j = 0; j < 5; j++) {
    float h = ab1[j];
    for (int i = 0; i < 26; i++) h += xw[i] * aw1[i * 5 + j];
    logit += fmaxf(h, 0.f) * aw2[j];
  }
  ws[OFF_LOGIT + (size_t)f * BB + b] = logit;
}

// ----------------------------------------- softmax over f + weighted pool
__global__ __launch_bounds__(256) void k_softmax_pool(
    const int* __restrict__ xcat, const float* __restrict__ emb_cat,
    const float* __restrict__ g_cat, const float* __restrict__ b_cat,
    const float* __restrict__ g_raw, const float* __restrict__ b_raw,
    const float* __restrict__ g_notu, const float* __restrict__ b_notu,
    float* __restrict__ ws) {
  int b = blockIdx.x * 256 + threadIdx.x;
  const float* lg = ws + OFF_LOGIT;
  float mx = -3.4e38f;
  for (int f = 0; f < NFEAT; f++) mx = fmaxf(mx, lg[(size_t)f * BB + b]);
  float s = 0;
  for (int f = 0; f < NFEAT; f++) s += expf(lg[(size_t)f * BB + b] - mx);
  float inv = 1.f / s;
  float acc[24];
  for (int c = 0; c < 24; c++) acc[c] = 0.f;
  for (int f = 0; f < NFEAT; f++) {
    float w = expf(lg[(size_t)f * BB + b] - mx) * inv;
    const float* chf = ws + ST_CH_FIN + f * 6;
    float mxc = chf[0], rsx = chf[1], mr = chf[2], rsr = chf[3], mn = chf[4], rsn = chf[5];
    int cat = xcat[b * (NFEAT + 1) + f];
    for (int l = 0; l < 4; l++)
      acc[l] += w * ((emb_cat[cat * 4 + l] - mxc) * rsx * g_cat[f] + b_cat[f]);
    for (int c = 0; c < 10; c++) {
      acc[4 + c]  += w * ((ws[OFF_RAW  + (size_t)(f * 10 + c) * BB + b] - mr) * rsr * g_raw[f]  + b_raw[f]);
      acc[14 + c] += w * ((ws[OFF_NOTU + (size_t)(f * 10 + c) * BB + b] - mn) * rsn * g_notu[f] + b_notu[f]);
    }
  }
  for (int c = 0; c < 24; c++) ws[OFF_POOL + (size_t)b * 24 + c] = acc[c];
}

// ------------------------------------- generic per-column mean/var sums
__global__ __launch_bounds__(256) void k_colstats(
    const float* __restrict__ x, int cols, float* __restrict__ sums, int rowsPerChunk) {
  int t = threadIdx.x;
  int c = blockIdx.x * 64 + (t & 63);
  float s = 0, ss = 0;
  if (c < cols) {
    int rend = (blockIdx.y + 1) * rowsPerChunk;
    for (int r = blockIdx.y * rowsPerChunk + (t >> 6); r < rend; r += 4) {
      float v = x[(size_t)r * cols + c];
      s += v; ss += v * v;
    }
  }
  __shared__ float red[256][2];
  red[t][0] = s; red[t][1] = ss;
  __syncthreads();
  if (t < 64 && c < cols) {
    float S  = red[t][0] + red[t + 64][0] + red[t + 128][0] + red[t + 192][0];
    float SS = red[t][1] + red[t + 64][1] + red[t + 128][1] + red[t + 192][1];
    atomicAdd(&sums[c * 2], S);
    atomicAdd(&sums[c * 2 + 1], SS);
  }
}

__global__ void k_finalize(const float* __restrict__ sums, float* __restrict__ fin,
                           int n, float count) {
  int i = blockIdx.x * 256 + threadIdx.x;
  if (i < n) {
    float m = sums[i * 2] / count;
    float v = sums[i * 2 + 1] / count - m * m;
    fin[i * 2] = m;
    fin[i * 2 + 1] = rsqrtf(v + EPS);
  }
}

// --------------------------- GEMM0: (B,24)@(24,1000), one wave per tile
__global__ __launch_bounds__(32) void k_gemm0(
    const float* __restrict__ w0, const float* __restrict__ bias0,
    const float* __restrict__ g_fin, const float* __restrict__ b_fin,
    float* __restrict__ ws) {
  int lane = threadIdx.x;
  int b0 = blockIdx.x * 16, n0 = blockIdx.y * 16;
  int hb = lane >> 4, lr = lane & 15;
  const float* pfin = ws + ST_PL_FIN;
  union { v16bf v; __bf16 h[16]; } A, Bf;
  for (int e = 0; e < 16; e++) {
    int k = hb * 8 + (e & 7) + (e >> 3) * 16;          // 0..31
    float a = 0.f;
    if (k < 24) {
      float p = ws[OFF_POOL + (size_t)(b0 + lr) * 24 + k];
      a = (p - pfin[k * 2]) * pfin[k * 2 + 1] * g_fin[k] + b_fin[k];
    }
    A.h[e] = (__bf16)a;
    int n = n0 + lr;
    float bb = (k < 24 && n < 1000) ? w0[k * 1000 + n] : 0.f;
    Bf.h[e] = (__bf16)bb;
  }
  v8f c;
  for (int i = 0; i < 8; i++) c[i] = 0.f;
  c = __builtin_amdgcn_wmma_f32_16x16x32_bf16(false, A.v, false, Bf.v, (short)0, c, false, false);
  float* y0 = ws + OFF_Y0;
  int n = n0 + lr;
  if (n < 1000) {
    float bias = bias0[n];
    for (int v = 0; v < 8; v++) {
      int row = b0 + v + hb * 8;
      y0[(size_t)row * 1000 + n] = fmaxf(c[v] + bias, 0.f);
    }
  }
}

// -------- apply y0 BN, emit zero-padded bf16 activation matrix (8192 x 1024)
__global__ __launch_bounds__(256) void k_bn_y0(
    const float* __restrict__ g, const float* __restrict__ bta,
    float* __restrict__ ws) {
  int idx = blockIdx.x * 256 + threadIdx.x;   // BB * 512 threads, 2 cols each
  int row = idx >> 9;
  int c2 = (idx & 511) * 2;
  const float* fin = ws + ST_Y0_FIN;
  const float* y0 = ws + OFF_Y0;
  float v0 = 0.f, v1 = 0.f;
  if (c2 < 1000) {
    float a = y0[(size_t)row * 1000 + c2];
    v0 = (a - fin[c2 * 2]) * fin[c2 * 2 + 1] * g[c2] + bta[c2];
    float b = y0[(size_t)row * 1000 + c2 + 1];
    v1 = (b - fin[(c2 + 1) * 2]) * fin[(c2 + 1) * 2 + 1] * g[c2 + 1] + bta[c2 + 1];
  }
  union { __bf16 h[2]; unsigned u; } p;
  p.h[0] = (__bf16)v0;
  p.h[1] = (__bf16)v1;
  ((unsigned*)(ws + OFF_Y0BF))[(size_t)row * 512 + (idx & 511)] = p.u;
}

// -------- pre-swizzle w1 into bf16 WMMA-fragment order: [kt][nt][lane][8dw]
__global__ __launch_bounds__(256) void k_swz_w1(
    const float* __restrict__ w1, float* __restrict__ ws) {
  int D = blockIdx.x * 256 + threadIdx.x;     // NTILES_K*NTILES_N*32*8 dwords
  int d = D & 7;
  int lane = (D >> 3) & 31;
  int tnt = D >> 8;
  int nt = tnt % NTILES_N;
  int kt = tnt / NTILES_N;
  int hb = lane >> 4, lr = lane & 15;
  int kk = hb * 8 + (d & 3) * 2 + (d >> 2) * 16;
  int k = kt * 32 + kk;
  int n = nt * 16 + lr;
  float lo = (k     < 1000 && n < 1500) ? w1[(size_t)k * 1500 + n]       : 0.f;
  float hi = (k + 1 < 1000 && n < 1500) ? w1[(size_t)(k + 1) * 1500 + n] : 0.f;
  union { __bf16 h[2]; unsigned u; } p;
  p.h[0] = (__bf16)lo;
  p.h[1] = (__bf16)hi;
  ((unsigned*)(ws + OFF_W1S))[D] = p.u;
}

// ----- TDM: async 2D tile load (64 rows x 32 bf16, row stride 1024 elems)
static __device__ __forceinline__ void tdm_load_a(unsigned ldsByteAddr,
                                                  unsigned long long gaddr) {
  uv4 g0;
  g0.x = 1u;                                        // count = 1 descriptor
  g0.y = ldsByteAddr;                               // LDS byte address
  g0.z = (unsigned)gaddr;                           // global addr [31:0]
  g0.w = ((unsigned)(gaddr >> 32) & 0x1FFFFFFu) | (2u << 30);  // addr hi | type=2
  uv8 g1;
  g1.s0 = 0x00010000u;                              // data_size = 2 bytes
  g1.s1 = (1024u & 0xFFFFu) << 16;                  // tensor_dim0 lo16 (=1024)
  g1.s2 = (1024u >> 16) | ((8192u & 0xFFFFu) << 16);// tensor_dim0 hi | dim1 lo (=8192)
  g1.s3 = (8192u >> 16) | (32u << 16);              // dim1 hi | tile_dim0 = 32
  g1.s4 = 64u;                                      // tile_dim1 = 64, tile_dim2 = 0
  g1.s5 = 1024u;                                    // tensor_dim0_stride lo32 (=1024)
  g1.s6 = 0u;
  g1.s7 = 0u;
  asm volatile("tensor_load_to_lds %0, %1" :: "s"(g0), "s"(g1) : "memory");
}

// -------- GEMM1: (B,1000)@(1000,1500) bf16 WMMA, TDM-fed A, preswizzled B
#define GM 64
#define GK 32
__global__ __launch_bounds__(256) void k_gemm1(
    const float* __restrict__ bias1, float* __restrict__ ws) {
  __shared__ __bf16 As[2][GM * GK];                 // 2 x 4 KB double buffer
  int t = threadIdx.x;
  int b0 = blockIdx.x * GM;
  int ntbase = blockIdx.y * 8;                      // 8 n-tiles (128 cols) per block
  int w = t >> 5, lane = t & 31, hb = lane >> 4, lr = lane & 15;
  int mi = w & 3, nset = (w >> 2) * 4;
  const uint4* w1s = (const uint4*)(ws + OFF_W1S);
  unsigned long long abase =
      (unsigned long long)(uintptr_t)(ws + OFF_Y0BF) + (unsigned long long)b0 * 2048ull;
  v8f acc[4];
  for (int j = 0; j < 4; j++)
    for (int i = 0; i < 8; i++) acc[j][i] = 0.f;

  if (w == 0) {                                     // prologue tile, buffer 0
    tdm_load_a(0u, abase);
    __builtin_amdgcn_s_wait_tensorcnt(0);
  }
  __syncthreads();

  for (int it = 0; it < NTILES_K; ++it) {
    int buf = it & 1;
    if (w == 0 && it + 1 < NTILES_K)                // prefetch next tile via TDM
      tdm_load_a((unsigned)((buf ^ 1) * (GM * GK * 2)),
                 abase + (unsigned long long)(it + 1) * GK * 2ull);

    union { v16bf v; unsigned u[8]; } a;
    const __bf16* Ab = &As[buf][0];
    int arow = (mi * 16 + lr) * GK;
    for (int d = 0; d < 8; d++) {
      int kk = hb * 8 + (d & 3) * 2 + (d >> 2) * 16;
      a.u[d] = *(const unsigned*)&Ab[arow + kk];
    }
    for (int jn = 0; jn < 4; jn++) {
      int nt = ntbase + nset + jn;
      union { v16bf v; uint4 q[2]; } bf;
      const uint4* p = w1s + (size_t)((it * NTILES_N + nt) * 32 + lane) * 2;
      bf.q[0] = p[0];
      bf.q[1] = p[1];
      acc[jn] = __builtin_amdgcn_wmma_f32_16x16x32_bf16(false, a.v, false, bf.v,
                                                        (short)0, acc[jn], false, false);
    }
    if (w == 0 && it + 1 < NTILES_K)
      __builtin_amdgcn_s_wait_tensorcnt(0);
    __syncthreads();
  }
  float* y1 = ws + OFF_Y1;
  for (int jn = 0; jn < 4; jn++) {
    int col = (ntbase + nset + jn) * 16 + lr;
    if (col < 1500) {
      float bias = bias1[col];
      for (int v = 0; v < 8; v++) {
        int row = b0 + mi * 16 + v + hb * 8;
        y1[(size_t)row * 1500 + col] = fmaxf(acc[jn][v] + bias, 0.f);
      }
    }
  }
}

// ------------------ GEMM2: (B,1500)@(1500,2) wave-per-row dot product
__global__ __launch_bounds__(256) void k_gemm2(
    const float* __restrict__ w2, const float* __restrict__ b2,
    const float* __restrict__ g_m2, const float* __restrict__ b_m2v,
    const float* __restrict__ ws, float* __restrict__ out) {
  __shared__ float W[3000];
  int t = threadIdx.x;
  for (int i = t; i < 3000; i += 256) W[i] = w2[i];
  __syncthreads();
  int wv = t >> 5, lane = t & 31;
  int b = blockIdx.x * 8 + wv;
  const float* y1 = ws + OFF_Y1 + (size_t)b * 1500;
  const float* fin = ws + ST_Y1_FIN;
  float a0 = 0, a1 = 0;
  for (int k = lane; k < 1500; k += 32) {
    float v = (y1[k] - fin[k * 2]) * fin[k * 2 + 1] * g_m2[k] + b_m2v[k];
    a0 += v * W[k * 2];
    a1 += v * W[k * 2 + 1];
  }
  a0 = warp_red(a0);
  a1 = warp_red(a1);
  if (lane == 0) {
    out[b * 2]     = a0 + b2[0];
    out[b * 2 + 1] = a1 + b2[1];
  }
}

// ================================================================ launch
extern "C" void kernel_launch(void* const* d_in, const int* in_sizes, int n_in,
                              void* d_out, int out_size, void* d_ws, size_t ws_size,
                              hipStream_t stream) {
  const int*   x_cat   = (const int*)d_in[0];
  const float* x_cont  = (const float*)d_in[1];
  const float* emb_cat = (const float*)d_in[2];
  const float* emb_feat  = (const float*)d_in[3];
  const float* emb_featw = (const float*)d_in[4];
  const float* raw_w1 = (const float*)d_in[5];
  const float* raw_b1 = (const float*)d_in[6];
  const float* raw_w2 = (const float*)d_in[7];
  const float* raw_b2 = (const float*)d_in[8];
  const float* notu_w1 = (const float*)d_in[9];
  const float* notu_b1 = (const float*)d_in[10];
  const float* notu_w2 = (const float*)d_in[11];
  const float* notu_b2 = (const float*)d_in[12];
  const float* att_w1 = (const float*)d_in[13];
  const float* att_b1 = (const float*)d_in[14];
  const float* att_w2 = (const float*)d_in[15];
  const float* att_b2 = (const float*)d_in[16];
  const float* g_cat = (const float*)d_in[17];
  const float* b_cat = (const float*)d_in[18];
  const float* g_raw = (const float*)d_in[19];
  const float* b_raw = (const float*)d_in[20];
  const float* g_notu = (const float*)d_in[21];
  const float* b_notu = (const float*)d_in[22];
  const float* g_feat = (const float*)d_in[23];
  const float* b_feat = (const float*)d_in[24];
  const float* g_featw = (const float*)d_in[25];
  const float* b_featw = (const float*)d_in[26];
  const float* g_w = (const float*)d_in[27];
  const float* b_w = (const float*)d_in[28];
  const float* g_fin = (const float*)d_in[29];
  const float* b_fin = (const float*)d_in[30];
  const float* mlp_w0 = (const float*)d_in[31];
  const float* mlp_b0 = (const float*)d_in[32];
  const float* g_m1 = (const float*)d_in[33];
  const float* b_m1 = (const float*)d_in[34];
  const float* mlp_w1 = (const float*)d_in[35];
  const float* mlp_b1 = (const float*)d_in[36];
  const float* g_m2 = (const float*)d_in[37];
  const float* b_m2 = (const float*)d_in[38];
  const float* mlp_w2 = (const float*)d_in[39];
  const float* mlp_b2 = (const float*)d_in[40];
  float* ws  = (float*)d_ws;
  float* out = (float*)d_out;

  k_zero<<<(STATS_N + 255) / 256, 256, 0, stream>>>(ws);
  k_feat_stats<<<1, 1024, 0, stream>>>(x_cat, emb_feat, emb_featw, ws);
  k_swz_w1<<<(NTILES_K * NTILES_N * 256) / 256, 256, 0, stream>>>(mlp_w1, ws);
  k_percol_mlp<<<dim3(BB / 256, NFEAT), 256, 0, stream>>>(
      x_cat, x_cont, emb_feat, emb_cat,
      raw_w1, raw_b1, raw_w2, raw_b2,
      notu_w1, notu_b1, notu_w2, notu_b2,
      g_feat, b_feat, ws);
  k_ch_final<<<1, 256, 0, stream>>>(ws);
  k_xw_sums<<<dim3(BB / 256, NFEAT), 256, 0, stream>>>(
      x_cat, emb_cat, emb_featw, g_cat, b_cat, g_featw, b_featw,
      g_raw, b_raw, g_notu, b_notu, ws);
  k_finalize<<<1, 256, 0, stream>>>(ws + ST_XW_SUM, ws + ST_XW_FIN, 26, (float)BB * NFEAT);
  k_logits<<<dim3(BB / 256, NFEAT), 256, 0, stream>>>(
      x_cat, emb_cat, emb_featw, g_cat, b_cat, g_featw, b_featw,
      g_raw, b_raw, g_notu, b_notu, g_w, b_w,
      att_w1, att_b1, att_w2, att_b2, ws);
  k_softmax_pool<<<BB / 256, 256, 0, stream>>>(
      x_cat, emb_cat, g_cat, b_cat, g_raw, b_raw, g_notu, b_notu, ws);
  k_colstats<<<dim3(1, 32), 256, 0, stream>>>(ws + OFF_POOL, 24, ws + ST_PL_SUM, BB / 32);
  k_finalize<<<1, 256, 0, stream>>>(ws + ST_PL_SUM, ws + ST_PL_FIN, 24, (float)BB);
  k_gemm0<<<dim3(BB / 16, (1000 + 15) / 16), 32, 0, stream>>>(mlp_w0, mlp_b0, g_fin, b_fin, ws);
  k_colstats<<<dim3((1000 + 63) / 64, 32), 256, 0, stream>>>(ws + OFF_Y0, 1000, ws + ST_Y0_SUM, BB / 32);
  k_finalize<<<(1000 + 255) / 256, 256, 0, stream>>>(ws + ST_Y0_SUM, ws + ST_Y0_FIN, 1000, (float)BB);
  k_bn_y0<<<(BB * 512) / 256, 256, 0, stream>>>(g_m1, b_m1, ws);
  k_gemm1<<<dim3(BB / GM, 12), 256, 0, stream>>>(mlp_b1, ws);
  k_colstats<<<dim3((1500 + 63) / 64, 32), 256, 0, stream>>>(ws + OFF_Y1, 1500, ws + ST_Y1_SUM, BB / 32);
  k_finalize<<<(1500 + 255) / 256, 256, 0, stream>>>(ws + ST_Y1_SUM, ws + ST_Y1_FIN, 1500, (float)BB);
  k_gemm2<<<BB / 8, 256, 0, stream>>>(mlp_w2, mlp_b2, g_m2, b_m2, ws, out);
}
